// GroupedQueryAttention_55052890800153
// MI455X (gfx1250) — matrix operand order repaired and checked by reading, not compiled, not run on hardware
//
#include <hip/hip_runtime.h>
#include <cstdint>
#include <cstddef>

// ---------------------------------------------------------------------------
// GQA attention for MI455X (gfx1250, wave32, WMMA bf16 16x16x32).
// Pipeline: cvt/transpose -> NT WMMA GEMMs (Q,K,V) -> RoPE(bf16) ->
//           flash-attention (TDM-staged K tile, LDS-transposed V, online
//           softmax) -> NT WMMA out-proj
// ---------------------------------------------------------------------------

typedef __attribute__((ext_vector_type(16))) __bf16 v16bf;
typedef __attribute__((ext_vector_type(8)))  float  v8f;
typedef __attribute__((ext_vector_type(4)))  unsigned int u32x4;
typedef __attribute__((ext_vector_type(8)))  int  i32x8;
typedef __attribute__((ext_vector_type(4)))  int  i32x4;

#if defined(__AMDGCN__) && __has_builtin(__builtin_amdgcn_tensor_load_to_lds)
#define ATTN_USE_TDM 1
#else
#define ATTN_USE_TDM 0
#endif

union Frag {
    v16bf  v;
    uint4  u[2];
    __bf16 h[16];
};

__device__ __forceinline__ v8f wmma_bf16f32(v16bf a, v16bf b, v8f c) {
    // D = A(16x32) * B(32x16) + C(16x16), f32 accumulate
    return __builtin_amdgcn_wmma_f32_16x16x32_bf16(
        /*neg_a=*/false, a, /*neg_b=*/false, b,
        /*c_mod=*/(short)0, c, /*reuse_a=*/false, /*reuse_b=*/false);
}

// --------------------------- elementwise kernels ---------------------------

__global__ void cvt_bf16_kernel(const float* __restrict__ x,
                                __bf16* __restrict__ y, size_t n) {
    size_t stride = (size_t)gridDim.x * blockDim.x;
    for (size_t i = (size_t)blockIdx.x * blockDim.x + threadIdx.x; i < n; i += stride)
        y[i] = (__bf16)x[i];
}

// src [R][C] row-major f32  ->  dst [C][R] row-major bf16 (weight transpose)
__global__ void transpose_cvt_kernel(const float* __restrict__ src,
                                     __bf16* __restrict__ dst, int R, int C) {
    size_t n = (size_t)R * C;
    size_t stride = (size_t)gridDim.x * blockDim.x;
    for (size_t i = (size_t)blockIdx.x * blockDim.x + threadIdx.x; i < n; i += stride) {
        int c = (int)(i % C);
        int r = (int)(i / C);
        dst[(size_t)c * R + r] = (__bf16)src[i];
    }
}

// RoPE (rotate-half) + bf16 convert + [B,S,h,D] -> [B,h,S,D] transpose.
// X: f32 [B*S, heads*D].  Y: bf16 [B, heads, S, D].  scale folded into output.
__global__ void rope_cvt_kernel(const float* __restrict__ X,
                                __bf16* __restrict__ Y,
                                int heads, float scale) {
    const int S = 2048, D = 128, B = 2;
    size_t total = (size_t)B * S * heads * (D / 2);
    size_t stride = (size_t)gridDim.x * blockDim.x;
    for (size_t i = (size_t)blockIdx.x * blockDim.x + threadIdx.x; i < total; i += stride) {
        int d = (int)(i % (D / 2));
        size_t t = i / (D / 2);
        int hh = (int)(t % heads); t /= heads;
        int s  = (int)(t % S);
        int b  = (int)(t / S);
        const float* src = X + ((size_t)(b * S + s) * heads + hh) * D;
        float x1 = src[d];
        float x2 = src[d + 64];
        // inv_freq = 10000^(-2d/128)
        float fr  = __powf(10000.0f, -(float)(2 * d) * (1.0f / 128.0f));
        float ang = (float)s * fr;
        float sn, cs;
        __sincosf(ang, &sn, &cs);
        float o1 = x1 * cs - x2 * sn;
        float o2 = x2 * cs + x1 * sn;
        __bf16* dst = Y + ((size_t)(b * heads + hh) * S + s) * D;
        dst[d]      = (__bf16)(o1 * scale);
        dst[d + 64] = (__bf16)(o2 * scale);
    }
}

// V: f32 [B*S, KV*D] -> bf16 [B, KV, S, D]
__global__ void vcvt_kernel(const float* __restrict__ X, __bf16* __restrict__ Y) {
    const int S = 2048, D = 128, B = 2, KV = 4;
    size_t total = (size_t)B * S * KV * D;
    size_t stride = (size_t)gridDim.x * blockDim.x;
    for (size_t i = (size_t)blockIdx.x * blockDim.x + threadIdx.x; i < total; i += stride) {
        int d = (int)(i % D);
        size_t t = i / D;
        int h = (int)(t % KV); t /= KV;
        int s = (int)(t % S);
        int b = (int)(t / S);
        Y[((size_t)(b * KV + h) * S + s) * D + d] =
            (__bf16)X[((size_t)(b * S + s) * KV + h) * D + d];
    }
}

// --------------------------- NT GEMM (WMMA bf16) ---------------------------
// C[M,N](f32) = A[M,K](bf16, row-major) * Bt[N,K](bf16, row-major = B^T)
// One wave computes a 32x32 tile (2x2 grid of 16x16 WMMA tiles).
__global__ void __launch_bounds__(128)
gemm_nt_bf16_kernel(const __bf16* __restrict__ A, const __bf16* __restrict__ Bt,
                    float* __restrict__ C, int M, int N, int K) {
    const int lane = threadIdx.x & 31;
    const int wid  = threadIdx.x >> 5;
    const int l15  = lane & 15;
    const int half = (lane >> 4) & 1;

    const int tilesN = N >> 5;
    int gw = blockIdx.x * 4 + wid;
    int mt = gw / tilesN;
    int nt = gw % tilesN;
    int m0 = mt * 32;
    int n0 = nt * 32;

    v8f a00 = {}, a01 = {}, a10 = {}, a11 = {};

    const __bf16* Ar0 = A  + (size_t)(m0 + l15)      * K;
    const __bf16* Ar1 = A  + (size_t)(m0 + 16 + l15) * K;
    const __bf16* Br0 = Bt + (size_t)(n0 + l15)      * K;
    const __bf16* Br1 = Bt + (size_t)(n0 + 16 + l15) * K;

    for (int kb = 0; kb < K; kb += 32) {
        Frag fa0, fa1, fb0, fb1;
        // A frag (16x32, 16-bit): 16 contiguous k at kb + half*16
        fa0.u[0] = *(const uint4*)(Ar0 + kb + half * 16);
        fa0.u[1] = *(const uint4*)(Ar0 + kb + half * 16 + 8);
        fa1.u[0] = *(const uint4*)(Ar1 + kb + half * 16);
        fa1.u[1] = *(const uint4*)(Ar1 + kb + half * 16 + 8);
        // B frag (32x16, 16-bit): 16 contiguous k at kb + half*16
        fb0.u[0] = *(const uint4*)(Br0 + kb + half * 16);
        fb0.u[1] = *(const uint4*)(Br0 + kb + half * 16 + 8);
        fb1.u[0] = *(const uint4*)(Br1 + kb + half * 16);
        fb1.u[1] = *(const uint4*)(Br1 + kb + half * 16 + 8);

        a00 = wmma_bf16f32(fa0.v, fb0.v, a00);
        a01 = wmma_bf16f32(fa0.v, fb1.v, a01);
        a10 = wmma_bf16f32(fa1.v, fb0.v, a10);
        a11 = wmma_bf16f32(fa1.v, fb1.v, a11);
    }

    // C/D layout: VGPR r -> row m = r + half*8; col n = l15
#pragma unroll
    for (int r = 0; r < 8; ++r) {
        size_t row0 = (size_t)(m0 + r + half * 8);
        C[row0 * N + n0 + l15]               = a00[r];
        C[row0 * N + n0 + 16 + l15]          = a01[r];
        C[(row0 + 16) * N + n0 + l15]        = a10[r];
        C[(row0 + 16) * N + n0 + 16 + l15]   = a11[r];
    }
}

// ------------------------- flash attention kernel --------------------------
// Q: bf16 [B,H,S,D] (pre-scaled by 1/sqrt(D)); Kg,Vg: bf16 [B,KV,S,D]
// O: bf16 [B,S,H*D].  Block = 128 threads (4 waves), covers 64 q rows.
// grid.x = B * H * (S/64)
__global__ void __launch_bounds__(128)
attn_kernel(const __bf16* __restrict__ Q, const __bf16* __restrict__ Kg,
            const __bf16* __restrict__ Vg, __bf16* __restrict__ O) {
    const int S = 2048, D = 128, H = 16, KV = 4;
    const int QT = S / 64;

    int blk = blockIdx.x;
    int qt  = blk % QT;
    int h   = (blk / QT) % H;
    int b   = blk / (QT * H);
    int kvh = h / (H / KV);

    const int lane = threadIdx.x & 31;
    const int wid  = threadIdx.x >> 5;
    const int l15  = lane & 15;
    const int half = (lane >> 4) & 1;
    const int q0   = qt * 64 + wid * 16;

    __shared__ __bf16 Kls[32 * 136];      // 32 keys x 128 d, row stride 136 (=256B+16B pad)
    __shared__ __bf16 Vtls[128 * 40];     // transposed: 128 d x 32 keys (+8 pad)
    __shared__ __bf16 Pls[4 * 16 * 40];   // per-wave P bounce: 16 q x 32 keys (+8 pad)

    // Load Q A-fragments once (4 chunks of K=32 covering D=128)
    Frag qf[4];
    const __bf16* Qrow = Q + ((size_t)(b * H + h) * S + q0 + l15) * D;
#pragma unroll
    for (int c = 0; c < 4; ++c) {
        qf[c].u[0] = *(const uint4*)(Qrow + c * 32 + half * 16);
        qf[c].u[1] = *(const uint4*)(Qrow + c * 32 + half * 16 + 8);
    }

    v8f o[8];
#pragma unroll
    for (int dc = 0; dc < 8; ++dc) o[dc] = (v8f){};
    float mrun[8], lrun[8];
#pragma unroll
    for (int r = 0; r < 8; ++r) { mrun[r] = -3.0e38f; lrun[r] = 0.0f; }

    const __bf16* Kbase = Kg + (size_t)(b * KV + kvh) * S * D;
    const __bf16* Vbase = Vg + (size_t)(b * KV + kvh) * S * D;
    __bf16* Pw = Pls + wid * 16 * 40;

#if ATTN_USE_TDM
    const unsigned ldsK = (unsigned)(unsigned long long)(const void*)&Kls[0];
#endif

    for (int kb = 0; kb < S; kb += 32) {
        __syncthreads();  // previous iteration's consumers done

        // Cooperative V staging: 32 keys x 128 d, transposed into Vtls[d][key]
#pragma unroll
        for (int i = 0; i < 2; ++i) {
            int c   = threadIdx.x + i * 128;      // 256 chunks of 16 bf16
            int key = c >> 3;
            int db  = (c & 7) * 16;
            Frag tv;
            const __bf16* vs = Vbase + (size_t)(kb + key) * D + db;
            tv.u[0] = *(const uint4*)vs;
            tv.u[1] = *(const uint4*)(vs + 8);
#pragma unroll
            for (int j = 0; j < 16; ++j)
                Vtls[(db + j) * 40 + key] = tv.h[j];
#if !ATTN_USE_TDM
            Frag tk;
            const __bf16* ks = Kbase + (size_t)(kb + key) * D + db;
            tk.u[0] = *(const uint4*)ks;
            tk.u[1] = *(const uint4*)(ks + 8);
            *(uint4*)&Kls[key * 136 + db]     = tk.u[0];
            *(uint4*)&Kls[key * 136 + db + 8] = tk.u[1];
#endif
            if (kb + 32 < S)  // prefetch next V tile -> global_prefetch_b8
                __builtin_prefetch((const void*)(vs + 32 * D), 0, 1);
        }

#if ATTN_USE_TDM
        // Tensor Data Mover: DMA the 32x128 bf16 K tile into Kls with LDS
        // padding 16B per 256B row (stride 272B == Kls row stride of 136 bf16).
        // 6-arg toolchain form: (u32x4 g0, i32x8 g1, i32x4 g2, i32x4 g3,
        //                        i32x8 g4, i32 cpol); groups 2+ unused (2D tile).
        if (wid == 0) {
            unsigned long long gK =
                (unsigned long long)(const void*)(Kbase + (size_t)kb * D);
            u32x4 g0;
            g0[0] = 1u;                                   // count=1 (valid user D#)
            g0[1] = ldsK;                                 // lds_addr (bytes)
            g0[2] = (unsigned)(gK & 0xffffffffu);         // global_addr[31:0]
            g0[3] = (unsigned)((gK >> 32) & 0x01ffffffu)  // global_addr[56:32]
                    | (2u << 30);                         // type = 2 ("image")
            i32x8 g1;
            g1[0] = (3 << 16)      // data_size = 3 (8-byte units)
                  | (1 << 20)      // pad_enable
                  | (5 << 22)      // pad_interval = 64 DWORDs (256 B)
                  | (3 << 25);     // pad_amount   = 4 DWORDs (16 B)
            g1[1] = (32 << 16);    // tensor_dim0 = 32 units (256 B row)
            g1[2] = (int)(2048u << 16);  // tensor_dim1[15:0] = 2048 rows
            g1[3] = (32 << 16);    // tile_dim0 = 32 units
            g1[4] = 32;            // tile_dim1 = 32 rows (tile_dim2 = 0)
            g1[5] = 32;            // tensor_dim0_stride = 32 units (contiguous)
            g1[6] = 0;
            g1[7] = 0;
            i32x4 gz4 = {};        // groups 2/3 unused (2D tensor)
            i32x8 gz8 = {};
            __builtin_amdgcn_tensor_load_to_lds(g0, g1, gz4, gz4, gz8, 0);
            __builtin_amdgcn_s_wait_tensorcnt(0);
        }
#endif
        __syncthreads();

        // ---- S = Q K^T over 32 keys (two 16-key column tiles) ----
        v8f s0 = (v8f){}, s1 = (v8f){};
#pragma unroll
        for (int c = 0; c < 4; ++c) {
            Frag k0, k1;
            const __bf16* kr0 = &Kls[(size_t)l15 * 136 + c * 32 + half * 16];
            const __bf16* kr1 = &Kls[(size_t)(16 + l15) * 136 + c * 32 + half * 16];
            k0.u[0] = *(const uint4*)kr0;  k0.u[1] = *(const uint4*)(kr0 + 8);
            k1.u[0] = *(const uint4*)kr1;  k1.u[1] = *(const uint4*)(kr1 + 8);
            s0 = wmma_bf16f32(qf[c].v, k0.v, s0);
            s1 = wmma_bf16f32(qf[c].v, k1.v, s1);
        }

        // ---- online softmax (row m = r + half*8; 16-lane butterfly) ----
#pragma unroll
        for (int r = 0; r < 8; ++r) {
            float t = fmaxf(s0[r], s1[r]);
            t = fmaxf(t, __shfl_xor(t, 1, 32));
            t = fmaxf(t, __shfl_xor(t, 2, 32));
            t = fmaxf(t, __shfl_xor(t, 4, 32));
            t = fmaxf(t, __shfl_xor(t, 8, 32));
            float mn = fmaxf(mrun[r], t);
            float al = __expf(mrun[r] - mn);
            mrun[r] = mn;
            float p0 = __expf(s0[r] - mn);
            float p1 = __expf(s1[r] - mn);
            float rs = p0 + p1;
            rs += __shfl_xor(rs, 1, 32);
            rs += __shfl_xor(rs, 2, 32);
            rs += __shfl_xor(rs, 4, 32);
            rs += __shfl_xor(rs, 8, 32);
            lrun[r] = lrun[r] * al + rs;
#pragma unroll
            for (int dc = 0; dc < 8; ++dc) o[dc][r] *= al;
            // stash P in per-wave LDS (C-layout -> memory), bf16
            Pw[(r + half * 8) * 40 + l15]      = (__bf16)p0;
            Pw[(r + half * 8) * 40 + 16 + l15] = (__bf16)p1;
        }

        // in-wave LDS RAW: stores above feed A-fragment loads below
        asm volatile("s_wait_dscnt 0" ::: "memory");

        // ---- O += P @ V ----
        Frag pf;
        pf.u[0] = *(const uint4*)&Pw[l15 * 40 + half * 16];
        pf.u[1] = *(const uint4*)&Pw[l15 * 40 + half * 16 + 8];
#pragma unroll
        for (int dc = 0; dc < 8; ++dc) {
            Frag vf;
            const __bf16* vr = &Vtls[(size_t)(dc * 16 + l15) * 40 + half * 16];
            vf.u[0] = *(const uint4*)vr;
            vf.u[1] = *(const uint4*)(vr + 8);
            o[dc] = wmma_bf16f32(pf.v, vf.v, o[dc]);
        }
    }

    // ---- epilogue: normalize, write bf16 [B,S,H*D] ----
#pragma unroll
    for (int r = 0; r < 8; ++r) {
        float inv = 1.0f / lrun[r];
        int srow = q0 + r + half * 8;
        __bf16* dst = O + ((size_t)(b * S + srow) * H + h) * D;
#pragma unroll
        for (int dc = 0; dc < 8; ++dc)
            dst[dc * 16 + l15] = (__bf16)(o[dc][r] * inv);
    }
}

// ------------------------------- launcher ----------------------------------

extern "C" void kernel_launch(void* const* d_in, const int* in_sizes, int n_in,
                              void* d_out, int out_size, void* d_ws, size_t ws_size,
                              hipStream_t stream) {
    (void)in_sizes; (void)n_in; (void)out_size; (void)ws_size;

    const float* x  = (const float*)d_in[0];
    const float* wq = (const float*)d_in[1];
    const float* wk = (const float*)d_in[2];
    const float* wv = (const float*)d_in[3];
    const float* wo = (const float*)d_in[4];
    float* out = (float*)d_out;

    const int B = 2, S = 2048, E = 2048, H = 16, KV = 4, D = 128;
    const size_t M  = (size_t)B * S;       // 4096
    const size_t HD = (size_t)H * D;       // 2048
    const size_t KD = (size_t)KV * D;      // 512

    // workspace carving (~108 MB)
    char* ws = (char*)d_ws;
    size_t off = 0;
    auto carve = [&](size_t bytes) -> void* {
        void* p = ws + off;
        off += (bytes + 255) & ~(size_t)255;
        return p;
    };
    __bf16* xb  = (__bf16*)carve(M * E * 2);         // x bf16 [M,E]
    __bf16* wqT = (__bf16*)carve(HD * E * 2);        // wq^T [HD,E]
    __bf16* wkT = (__bf16*)carve(KD * E * 2);        // wk^T [KD,E]
    __bf16* wvT = (__bf16*)carve(KD * E * 2);        // wv^T [KD,E]
    __bf16* woT = (__bf16*)carve(E * HD * 2);        // wo^T [E,HD]
    float*  Qf  = (float*)carve(M * HD * 4);         // Q f32 [M,HD]
    float*  Kf  = (float*)carve(M * KD * 4);         // K f32 [M,KD]
    float*  Vf  = (float*)carve(M * KD * 4);         // V f32 [M,KD]
    __bf16* Qb  = (__bf16*)carve(M * HD * 2);        // Q rope bf16 [B,H,S,D]
    __bf16* Kb  = (__bf16*)carve(M * KD * 2);        // K rope bf16 [B,KV,S,D]
    __bf16* Vb  = (__bf16*)carve(M * KD * 2);        // V bf16 [B,KV,S,D]
    __bf16* Ob  = (__bf16*)Qf;                       // alias: Qf dead after rope

    const float scale = 0.08838834764831845f;        // 1/sqrt(128)

    // 1) conversions / weight transposes
    cvt_bf16_kernel<<<4096, 256, 0, stream>>>(x, xb, M * E);
    transpose_cvt_kernel<<<2048, 256, 0, stream>>>(wq, wqT, E, (int)HD);
    transpose_cvt_kernel<<<1024, 256, 0, stream>>>(wk, wkT, E, (int)KD);
    transpose_cvt_kernel<<<1024, 256, 0, stream>>>(wv, wvT, E, (int)KD);
    transpose_cvt_kernel<<<2048, 256, 0, stream>>>(wo, woT, (int)HD, E);

    // 2) projections (NT WMMA GEMM): grid = (M/32)*(N/32)/4 blocks of 128
    gemm_nt_bf16_kernel<<<(int)((M / 32) * (HD / 32) / 4), 128, 0, stream>>>(
        xb, wqT, Qf, (int)M, (int)HD, E);
    gemm_nt_bf16_kernel<<<(int)((M / 32) * (KD / 32) / 4), 128, 0, stream>>>(
        xb, wkT, Kf, (int)M, (int)KD, E);
    gemm_nt_bf16_kernel<<<(int)((M / 32) * (KD / 32) / 4), 128, 0, stream>>>(
        xb, wvT, Vf, (int)M, (int)KD, E);

    // 3) RoPE (+ head-major transpose, + fold 1/sqrt(D) into Q) and V convert
    rope_cvt_kernel<<<4096, 256, 0, stream>>>(Qf, Qb, H, scale);
    rope_cvt_kernel<<<2048, 256, 0, stream>>>(Kf, Kb, KV, 1.0f);
    vcvt_kernel<<<2048, 256, 0, stream>>>(Vf, Vb);

    // 4) flash attention: grid = B*H*(S/64)
    attn_kernel<<<B * H * (S / 64), 128, 0, stream>>>(Qb, Kb, Vb, Ob);

    // 5) output projection -> f32 d_out [B*S, E]
    gemm_nt_bf16_kernel<<<(int)((M / 32) * (E / 32) / 4), 128, 0, stream>>>(
        Ob, woT, out, (int)M, E, (int)HD);
}
